// MultiHeadAttention_tem_bias_59502476919153
// MI455X (gfx1250) — matrix operand order repaired and checked by reading, not compiled, not run on hardware
//
#include <hip/hip_runtime.h>

typedef __attribute__((ext_vector_type(16))) _Float16 v16h;
typedef __attribute__((ext_vector_type(8)))  _Float16 v8h;
typedef __attribute__((ext_vector_type(8)))  float    v8f;
typedef __attribute__((ext_vector_type(4)))  float    v4f;

#define B_  4
#define L_  2048
#define D_  512
#define H_  8
#define DK_ 64

// D = A(16x32 f16) * B(32x16 f16) + C (f32 accum), wave32 WMMA
__device__ __forceinline__ v8f wmma_f16(v16h a, v16h b, v8f c) {
  return __builtin_amdgcn_wmma_f32_16x16x32_f16(false, a, false, b, (short)0, c, false, false);
}

// A-matrix per-lane layout (matched to B): lanes 0-15 hold K 0..15,
// lanes 16-31 hold K 16..31 -> two contiguous 8-half chunks.
__device__ __forceinline__ v16h load_a_f16(const _Float16* p, int sub) {
  v16h r;
  v8h lo = *(const v8h*)(p + sub * 16);
  v8h hi = *(const v8h*)(p + 8 + sub * 16);
#pragma unroll
  for (int i = 0; i < 8; i++) { r[i] = lo[i]; r[i + 8] = hi[i]; }
  return r;
}
__device__ __forceinline__ v16h cvt_a_f32(const float* p, int sub) {
  v16h r;
#pragma unroll
  for (int i = 0; i < 8; i++) {
    r[i]     = (_Float16)p[sub * 16 + i];
    r[i + 8] = (_Float16)p[8 + sub * 16 + i];
  }
  return r;
}
// B-matrix per-lane layout: lane = column N, contiguous 16 halves along K.
__device__ __forceinline__ v16h load_b_f16(const _Float16* p, int sub) {
  return *(const v16h*)(p + sub * 16);
}
__device__ __forceinline__ v16h cvt_b_f32(const float* p, int sub) {
  v16h r;
#pragma unroll
  for (int i = 0; i < 16; i++) r[i] = (_Float16)p[sub * 16 + i];
  return r;
}

// C = A[8192,512](f32) @ W[512,512]^T (f32) -> f16, stored head-split.
// transposeStore=0: dst[b][h][l][d]   (row-major per head, for Q/K)
// transposeStore=1: dst[b][h][d][l]   (transposed, for V -> contiguous B-frags in P*V)
__global__ __launch_bounds__(32)
void proj_kernel(const float* __restrict__ A, const float* __restrict__ W,
                 _Float16* __restrict__ dst, int transposeStore) {
  const int lane = threadIdx.x;
  const int sub = lane >> 4, lm = lane & 15;
  const int col0 = blockIdx.x * 16;
  const int row0 = blockIdx.y * 16;
  const float* ap = A + (size_t)(row0 + lm) * D_;
  const float* wp = W + (size_t)(col0 + lm) * D_;
  v8f acc = {};
#pragma unroll 4
  for (int kk = 0; kk < D_; kk += 32) {
    v16h a = cvt_a_f32(ap + kk, sub);
    v16h b = cvt_b_f32(wp + kk, sub);
    acc = wmma_f16(a, b, acc);
  }
  const int n = col0 + lm;
  const int h = n >> 6, d = n & 63;
#pragma unroll
  for (int r = 0; r < 8; r++) {
    int m = sub * 8 + r;               // C/D layout: VGPR r, lanes 0-15 -> M=r, 16-31 -> M=8+r
    int grow = row0 + m;
    int bb = grow >> 11, l = grow & 2047;
    size_t idx = transposeStore
      ? (((size_t)(bb * H_ + h) * DK_ + d) * L_ + l)
      : (((size_t)(bb * H_ + h) * L_ + l) * DK_ + d);
    dst[idx] = (_Float16)acc[r];
  }
}

// One block = one (b,h, 16-query tile). Scores for all 2048 keys in 128KB LDS,
// softmax in LDS, single non-temporal attn write, then P@V via WMMA from LDS.
__global__ __launch_bounds__(128)
void attn_kernel(const _Float16* __restrict__ qh, const _Float16* __restrict__ kh,
                 const _Float16* __restrict__ vhT, const float* __restrict__ q,
                 const float* __restrict__ Wb, float* __restrict__ attn,
                 _Float16* __restrict__ ctx) {
  __shared__ float sS[16 * L_];                 // 128 KB (CDNA5 WGP LDS = 320 KB)
  __shared__ float sBias[16], sMax[16], sSum[16];

  const int t  = threadIdx.x;
  const int bi = blockIdx.x;
  const int qt = bi & 127;
  const int bh = bi >> 7;
  const int h = bh & 7, b = bh >> 3;
  const int qbase = qt * 16;

  // per-query-row temporal bias: dot(q[b,row,:], Wb[h,:]) with 8 lanes/row
  {
    int row = t >> 3, s8 = t & 7;
    const float* qp = q + ((size_t)b * L_ + qbase + row) * D_;
    const float* wp = Wb + (size_t)h * D_;
    float acc = 0.f;
    for (int j = s8; j < D_; j += 8) acc += qp[j] * wp[j];
    acc += __shfl_xor(acc, 1);
    acc += __shfl_xor(acc, 2);
    acc += __shfl_xor(acc, 4);
    if (s8 == 0) sBias[row] = acc;
  }
  __syncthreads();

  const int wave = t >> 5, lane = t & 31, sub = lane >> 4, lm = lane & 15;

  // phase 1: scores = (qh . kh^T)/sqrt(DK) + bias  -> LDS
  const _Float16* qrow = qh + (((size_t)bh * L_) + qbase + lm) * DK_;
  v16h a0 = load_a_f16(qrow, sub);              // K 0..31
  v16h a1 = load_a_f16(qrow + 32, sub);         // K 32..63
  float fbias[8];
#pragma unroll
  for (int r = 0; r < 8; r++) fbias[r] = sBias[sub * 8 + r];
  for (int nt = wave; nt < 128; nt += 4) {      // 128 key tiles of 16, 4 waves
    const _Float16* kb = kh + (((size_t)bh * L_) + nt * 16 + lm) * DK_;
    v16h b0 = load_b_f16(kb, sub);
    v16h b1 = load_b_f16(kb + 32, sub);
    v8f acc = {};
    acc = wmma_f16(a0, b0, acc);
    acc = wmma_f16(a1, b1, acc);
#pragma unroll
    for (int r = 0; r < 8; r++) {
      int m = sub * 8 + r;
      sS[m * L_ + nt * 16 + lm] = acc[r] * 0.125f + fbias[r];
    }
  }
  __syncthreads();

  // phase 2: row max + sumexp (8 lanes per row, shuffle reduce)
  {
    int row = t >> 3, s8 = t & 7;
    const float* rp = sS + row * L_ + s8 * 256;
    float mx = -3.0e38f;
    for (int j = 0; j < 256; j++) mx = fmaxf(mx, rp[j]);
    mx = fmaxf(mx, __shfl_xor(mx, 1));
    mx = fmaxf(mx, __shfl_xor(mx, 2));
    mx = fmaxf(mx, __shfl_xor(mx, 4));
    float sm = 0.f;
    for (int j = 0; j < 256; j++) sm += __expf(rp[j] - mx);
    sm += __shfl_xor(sm, 1);
    sm += __shfl_xor(sm, 2);
    sm += __shfl_xor(sm, 4);
    if (s8 == 0) { sMax[row] = mx; sSum[row] = sm; }
  }
  __syncthreads();

  // normalize in LDS + single attn write (the bandwidth floor).
  // float4 lanes + non-temporal stores: attn is write-once/read-never, so NT
  // keeps the f16 K/V/ctx working set resident in the 192MB L2.
  for (int r = 0; r < 16; r++) {
    float rm = sMax[r];
    float ri = 1.0f / sSum[r];
    float* rowp = sS + r * L_;
    float* gout = attn + (((size_t)bh * L_) + qbase + r) * L_;
#pragma unroll
    for (int jj = 0; jj < L_ / 512; jj++) {     // 4 iterations, 4 floats/lane
      int j = jj * 512 + t * 4;
      v4f pv = *(const v4f*)(rowp + j);
      pv.x = __expf(pv.x - rm) * ri;
      pv.y = __expf(pv.y - rm) * ri;
      pv.z = __expf(pv.z - rm) * ri;
      pv.w = __expf(pv.w - rm) * ri;
      *(v4f*)(rowp + j) = pv;
      __builtin_nontemporal_store(pv, (v4f*)(gout + j));
    }
  }
  __syncthreads();

  // phase 3: ctx tile = P[16,2048] @ V[2048,64]; wave w owns DV columns [w*16, w*16+16)
  {
    const _Float16* vb = vhT + (((size_t)bh * DK_) + wave * 16 + lm) * L_;
    const float* prow = sS + lm * L_;
    v8f acc = {};
    for (int kk = 0; kk < L_; kk += 32) {
      v16h a  = cvt_a_f32(prow + kk, sub);      // P from LDS, cvt f32->f16
      v16h bf = load_b_f16(vb + kk, sub);       // contiguous thanks to V^T layout
      acc = wmma_f16(a, bf, acc);
    }
#pragma unroll
    for (int r = 0; r < 8; r++) {
      int m = sub * 8 + r;
      ctx[((size_t)b * L_ + qbase + m) * (H_ * DK_) + h * DK_ + wave * 16 + lm] =
          (_Float16)acc[r];
    }
  }
}

// out = ctx[8192,512](f16) @ Wfc[512,512]^T + bfc  -> f32
__global__ __launch_bounds__(32)
void fc_kernel(const _Float16* __restrict__ ctx, const float* __restrict__ Wfc,
               const float* __restrict__ bfc, float* __restrict__ out) {
  const int lane = threadIdx.x;
  const int sub = lane >> 4, lm = lane & 15;
  const int col0 = blockIdx.x * 16, row0 = blockIdx.y * 16;
  const _Float16* ap = ctx + (size_t)(row0 + lm) * D_;
  const float* wp = Wfc + (size_t)(col0 + lm) * D_;
  v8f acc = {};
#pragma unroll 4
  for (int kk = 0; kk < D_; kk += 32) {
    v16h a = load_a_f16(ap + kk, sub);
    v16h b = cvt_b_f32(wp + kk, sub);
    acc = wmma_f16(a, b, acc);
  }
  float bias = bfc[col0 + lm];
#pragma unroll
  for (int r = 0; r < 8; r++) {
    int m = sub * 8 + r;
    out[(size_t)(row0 + m) * D_ + col0 + lm] = acc[r] + bias;
  }
}

extern "C" void kernel_launch(void* const* d_in, const int* in_sizes, int n_in,
                              void* d_out, int out_size, void* d_ws, size_t ws_size,
                              hipStream_t stream) {
  (void)in_sizes; (void)n_in; (void)out_size; (void)ws_size;
  const float* q   = (const float*)d_in[0];
  const float* k   = (const float*)d_in[1];
  const float* v   = (const float*)d_in[2];
  const float* Wq  = (const float*)d_in[3];
  const float* Wk  = (const float*)d_in[4];
  const float* Wv  = (const float*)d_in[5];
  const float* Wb  = (const float*)d_in[6];
  const float* Wfc = (const float*)d_in[7];
  const float* bfc = (const float*)d_in[8];

  float* out  = (float*)d_out;
  float* attn = out + (size_t)B_ * L_ * D_;     // outputs concatenated: out, then attn

  const size_t nHead = (size_t)B_ * H_ * L_ * DK_;   // 4,194,304 halves = 8 MB each
  _Float16* qh  = (_Float16*)d_ws;
  _Float16* kh  = qh + nHead;
  _Float16* vhT = kh + nHead;
  _Float16* ctx = vhT + nHead;                  // total ws use: 32 MB

  dim3 pgrid(D_ / 16, (B_ * L_) / 16);          // 32 x 512 tiles, one wave each
  proj_kernel<<<pgrid, 32, 0, stream>>>(q, Wq, qh, 0);
  proj_kernel<<<pgrid, 32, 0, stream>>>(k, Wk, kh, 0);
  proj_kernel<<<pgrid, 32, 0, stream>>>(v, Wv, vhT, 1);

  attn_kernel<<<B_ * H_ * (L_ / 16), 128, 0, stream>>>(qh, kh, vhT, q, Wb, attn, ctx);

  fc_kernel<<<pgrid, 32, 0, stream>>>(ctx, Wfc, bfc, out);
}
